// CrossAttention_7499012898971
// MI455X (gfx1250) — matrix operand order repaired and checked by reading, not compiled, not run on hardware
//
#include <hip/hip_runtime.h>
#include <hip/hip_bf16.h>
#include <math.h>
#include <stdint.h>

typedef __attribute__((ext_vector_type(16))) __bf16 v16bf;
typedef __attribute__((ext_vector_type(8)))  __bf16 v8bf;
typedef __attribute__((ext_vector_type(4)))  __bf16 v4bf;
typedef __attribute__((ext_vector_type(8)))  float  v8f;
typedef __attribute__((ext_vector_type(4)))  unsigned int u32x4;
typedef __attribute__((ext_vector_type(8)))  int i32x8;
typedef __attribute__((ext_vector_type(4)))  int i32x4;

#define C_DIM   256
#define N_QKV   768
#define H_HEADS 8
#define D_HEAD  32
#define L_SEQ   20480
#define M_ROWS  81920
#define NSEG    20            /* L_SEQ split into 20 segments of 1024 rows */
#define EPS_F   1e-6f

// ---------------------------------------------------------------------------
// A-fragment: 16x32 bf16, row-major source with leading dim C_DIM.
// ISA layout: lanes 0-15 hold M=lane with K = {0..7, 16..23}; lanes 16-31
// hold M=lane-16 with K = {8..15, 24..31}.  -> two contiguous b128 loads.
// ---------------------------------------------------------------------------
__device__ inline v16bf load_a_frag(const __bf16* __restrict__ A, int rowBase,
                                    int k0, int lane) {
    int row = rowBase + (lane & 15);
    int kb  = k0 + ((lane >> 4) << 3);
    const v8bf lo = *reinterpret_cast<const v8bf*>(A + (size_t)row * C_DIM + kb);
    const v8bf hi = *reinterpret_cast<const v8bf*>(A + (size_t)row * C_DIM + kb + 16);
    return __builtin_shufflevector(lo, hi, 0,1,2,3,4,5,6,7,8,9,10,11,12,13,14,15);
}

// ---------------------------------------------------------------------------
// B-fragment: 32x16 bf16, pre-packed so each lane reads 16 contiguous bf16.
// Packed order: [((nt * ntk + kt) * 32 + lane) * 16 + e]
// with lane -> N = lane&15, element e -> K = (e&7) + 8*(lane>=16) + 16*(e>=8).
// ---------------------------------------------------------------------------
__device__ inline v16bf load_b_frag(const __bf16* __restrict__ Bp, int ntk,
                                    int nt, int kt, int lane) {
    return *reinterpret_cast<const v16bf*>(
        Bp + ((size_t)(nt * ntk + kt) * 32 + lane) * 16);
}

// ---------------------------------------------------------------------------
// Tensor Data Mover: async 2D tile (rows x width bf16, row stride in elems)
// from global to LDS.  D# per CDNA5 ISA ch.8:
//   group0: [1:0]=count=1, [63:32]=lds_addr, [120:64]=global_addr, [127:126]=2
//   group1: [17:16]=data_size(2B), tensor_dim0/1, tile_dim0/1, dim0_stride
// Issued by one full wave; tracked by TENSORcnt.
// ---------------------------------------------------------------------------
#if __has_builtin(__builtin_amdgcn_tensor_load_to_lds)
#define HAVE_TDM 1
__device__ inline void tdm_load_tile_2d(unsigned lds_off, const void* gptr,
                                        unsigned width, unsigned rows,
                                        unsigned row_stride) {
    unsigned long long ga = (unsigned long long)(uintptr_t)gptr;
    u32x4 g0;
    g0.x = 1u;                                   /* count=1, user mode */
    g0.y = lds_off;                              /* lds_addr            */
    g0.z = (unsigned)(ga & 0xffffffffu);         /* global_addr[31:0]   */
    g0.w = (unsigned)((ga >> 32) & 0x1ffffffu)   /* global_addr[56:32]  */
         | (2u << 30);                           /* type = 2 (image)    */
    i32x8 g1;
    g1[0] = (int)(1u << 16);                     /* data_size = 2 bytes */
    g1[1] = (int)(width << 16);                  /* tensor_dim0 lo16    */
    g1[2] = (int)(rows << 16);                   /* tensor_dim1 lo16    */
    g1[3] = (int)(width << 16);                  /* tile_dim0           */
    g1[4] = (int)rows;                           /* tile_dim1           */
    g1[5] = (int)row_stride;                     /* dim0_stride lo32    */
    g1[6] = 0;
    g1[7] = 0;
    i32x4 z4 = {0, 0, 0, 0};
#if defined(__clang_major__) && __clang_major__ >= 23
    i32x8 z8 = {0, 0, 0, 0, 0, 0, 0, 0};
    __builtin_amdgcn_tensor_load_to_lds(g0, g1, z4, z4, z8, 0);
#else
    __builtin_amdgcn_tensor_load_to_lds(g0, g1, z4, z4, 0);
#endif
}
#else
#define HAVE_TDM 0
#endif

// ---------------------------------------------------------------------------
// x (f32) -> bf16
// ---------------------------------------------------------------------------
__global__ __launch_bounds__(256) void f32_to_bf16(const float* __restrict__ src,
                                                   __bf16* __restrict__ dst, int n) {
    int i = (blockIdx.x * 256 + threadIdx.x) * 4;
    if (i >= n) return;
    float4 f = *reinterpret_cast<const float4*>(src + i);
    v4bf o;
    o.x = (__bf16)f.x; o.y = (__bf16)f.y; o.z = (__bf16)f.z; o.w = (__bf16)f.w;
    *reinterpret_cast<v4bf*>(dst + i) = o;
}

// ---------------------------------------------------------------------------
// Pack a row-major f32 weight [K x N] into bf16 WMMA B-fragment order.
// ---------------------------------------------------------------------------
__global__ __launch_bounds__(256) void pack_b(const float* __restrict__ W,
                                              __bf16* __restrict__ Bp,
                                              int Kdim, int Ndim) {
    int p = blockIdx.x * 256 + threadIdx.x;
    if (p >= Kdim * Ndim) return;
    int e    = p & 15;
    int lane = (p >> 4) & 31;
    int t    = p >> 9;
    int ntk  = Kdim >> 5;
    int kt   = t % ntk;
    int nt   = t / ntk;
    int n    = nt * 16 + (lane & 15);
    int kk   = kt * 32 + (e & 7) + ((lane >> 4) << 3) + ((e >> 3) << 4);
    Bp[p] = (__bf16)W[(size_t)kk * Ndim + n];
}

// ---------------------------------------------------------------------------
// GEMM1: qkv = x @ W_qkv, epilogue splits q/k/v and applies elu+1 to q,k.
// Block tile 128x64, 8 waves, each wave a 32x32 tile = 2x2 WMMA accumulators.
// ---------------------------------------------------------------------------
__global__ __launch_bounds__(256) void qkv_gemm(const __bf16* __restrict__ X,
                                                const __bf16* __restrict__ Wp,
                                                __bf16* __restrict__ Q,
                                                __bf16* __restrict__ K,
                                                __bf16* __restrict__ Vv) {
    int tid  = threadIdx.x;
    int lane = tid & 31;
    int wave = tid >> 5;
    int wm   = wave & 3;
    int wn   = wave >> 2;
    int m0   = blockIdx.x * 128 + wm * 32;
    int n0   = blockIdx.y * 64  + wn * 32;
    int nt   = n0 >> 4;

    v8f acc[2][2] = {};
#pragma unroll
    for (int kt = 0; kt < 8; ++kt) {
        v16bf a0 = load_a_frag(X, m0,      kt * 32, lane);
        v16bf a1 = load_a_frag(X, m0 + 16, kt * 32, lane);
        v16bf b0 = load_b_frag(Wp, 8, nt,     kt, lane);
        v16bf b1 = load_b_frag(Wp, 8, nt + 1, kt, lane);
        acc[0][0] = __builtin_amdgcn_wmma_f32_16x16x32_bf16(false, a0, false, b0,
                        (short)0, acc[0][0], false, false);
        acc[0][1] = __builtin_amdgcn_wmma_f32_16x16x32_bf16(false, a0, false, b1,
                        (short)0, acc[0][1], false, false);
        acc[1][0] = __builtin_amdgcn_wmma_f32_16x16x32_bf16(false, a1, false, b0,
                        (short)0, acc[1][0], false, false);
        acc[1][1] = __builtin_amdgcn_wmma_f32_16x16x32_bf16(false, a1, false, b1,
                        (short)0, acc[1][1], false, false);
    }

#pragma unroll
    for (int i = 0; i < 2; ++i) {
#pragma unroll
        for (int j = 0; j < 2; ++j) {
            int col     = n0 + j * 16 + (lane & 15);   // 0..767
            int s       = col >> 8;                    // 0=q 1=k 2=v
            int c       = col & 255;
            int rowBase = m0 + i * 16 + ((lane >> 4) << 3);
            __bf16* dst = (s == 0) ? Q : (s == 1) ? K : Vv;
#pragma unroll
            for (int e = 0; e < 8; ++e) {
                float val = acc[i][j][e];
                if (s < 2) val = (val > 0.0f) ? (val + 1.0f) : __expf(val);
                dst[(size_t)(rowBase + e) * C_DIM + c] = (__bf16)val;
            }
        }
    }
}

// ---------------------------------------------------------------------------
// Stage A: partial kv/ksum over a 1024-row segment of L.
// Grid (NSEG, 32) = 640 blocks so the 84 MB k/v stream can use the whole chip.
// Tiles staged into LDS by the Tensor Data Mover when available.
// ---------------------------------------------------------------------------
__global__ __launch_bounds__(256) void kv_partial(const __bf16* __restrict__ K,
                                                  const __bf16* __restrict__ Vv,
                                                  float* __restrict__ KVp,
                                                  float* __restrict__ KSp) {
    __shared__ __bf16 lk[128 * 32];
    __shared__ __bf16 lv[128 * 32];
    int seg = blockIdx.x;          // 0..NSEG-1
    int nh  = blockIdx.y;          // 0..31
    int n   = nh >> 3;
    int h   = nh & 7;
    int tid = threadIdx.x;
    int m   = tid >> 3;            // output row (0..31)
    int d0  = (tid & 7) * 4;       // 4 consecutive d per thread

    float acc0 = 0.f, acc1 = 0.f, acc2 = 0.f, acc3 = 0.f;
    float ks   = 0.f;
    size_t base = ((size_t)n * L_SEQ + (size_t)seg * 1024) * C_DIM + h * 32;

    for (int c = 0; c < 8; ++c) {
        size_t off = base + (size_t)c * 128 * C_DIM;
#if HAVE_TDM
        if (tid < 32) {                       // wave 0 issues the DMAs
            tdm_load_tile_2d((unsigned)(uintptr_t)&lk[0], K + off, 32, 128, C_DIM);
            tdm_load_tile_2d((unsigned)(uintptr_t)&lv[0], Vv + off, 32, 128, C_DIM);
            __builtin_amdgcn_s_wait_tensorcnt(0);
        }
#else
        for (int i = tid; i < 128 * 32; i += 256) {
            int r = i >> 5, cc = i & 31;
            lk[i] = K[off + (size_t)r * C_DIM + cc];
            lv[i] = Vv[off + (size_t)r * C_DIM + cc];
        }
#endif
        __syncthreads();
#pragma unroll 4
        for (int r = 0; r < 128; ++r) {
            float vm = (float)lv[r * 32 + m];
            acc0 += vm * (float)lk[r * 32 + d0];
            acc1 += vm * (float)lk[r * 32 + d0 + 1];
            acc2 += vm * (float)lk[r * 32 + d0 + 2];
            acc3 += vm * (float)lk[r * 32 + d0 + 3];
        }
        if (tid < 32) {
#pragma unroll 4
            for (int r = 0; r < 128; ++r) ks += (float)lk[r * 32 + tid];
        }
        __syncthreads();
    }
    float4 o = make_float4(acc0, acc1, acc2, acc3);
    *reinterpret_cast<float4*>(KVp + ((size_t)(nh * NSEG + seg)) * 1024 + tid * 4) = o;
    if (tid < 32) KSp[(nh * NSEG + seg) * 32 + tid] = ks;
}

// ---------------------------------------------------------------------------
// Stage B: combine NSEG partials (deterministic in-order sum, no atomics).
// ---------------------------------------------------------------------------
__global__ __launch_bounds__(256) void kv_finalize(const float* __restrict__ KVp,
                                                   const float* __restrict__ KSp,
                                                   float* __restrict__ KV,
                                                   float* __restrict__ KS) {
    int nh  = blockIdx.x;
    int tid = threadIdx.x;
    float4 acc = make_float4(0.f, 0.f, 0.f, 0.f);
    for (int s = 0; s < NSEG; ++s) {
        float4 p = *reinterpret_cast<const float4*>(
            KVp + ((size_t)(nh * NSEG + s)) * 1024 + tid * 4);
        acc.x += p.x; acc.y += p.y; acc.z += p.z; acc.w += p.w;
    }
    *reinterpret_cast<float4*>(KV + (size_t)nh * 1024 + tid * 4) = acc;
    if (tid < 32) {
        float s = 0.f;
        for (int q = 0; q < NSEG; ++q) s += KSp[(nh * NSEG + q) * 32 + tid];
        KS[nh * 32 + tid] = s;
    }
}

// ---------------------------------------------------------------------------
// y[l,h,m] = (1/(q.ksum+eps)) * sum_d q[l,h,d]*kv[n,h,m,d]
// One wave per (token,head); lane owns m.
// ---------------------------------------------------------------------------
__global__ __launch_bounds__(256) void y_kernel(const __bf16* __restrict__ Q,
                                                const float* __restrict__ KV,
                                                const float* __restrict__ KS,
                                                __bf16* __restrict__ Y) {
    int wave = threadIdx.x >> 5;
    int lane = threadIdx.x & 31;
    int idx  = blockIdx.x * 8 + wave;   // (row,head) pair
    int row  = idx >> 3;
    int h    = idx & 7;
    int n    = row / L_SEQ;
    int nh   = n * 8 + h;

    const __bf16* qrow = Q + (size_t)row * C_DIM + h * 32;
    float qv[32];
#pragma unroll
    for (int i = 0; i < 32; ++i) qv[i] = (float)qrow[i];

    const float* ks = KS + (size_t)nh * 32;
    float denom = EPS_F;
#pragma unroll
    for (int i = 0; i < 32; ++i) denom += qv[i] * ks[i];
    float z = 1.0f / denom;

    const float* kvr = KV + ((size_t)nh * 32 + lane) * 32;
    float acc = 0.f;
#pragma unroll
    for (int i = 0; i < 32; ++i) acc += qv[i] * kvr[i];

    Y[(size_t)row * C_DIM + h * 32 + lane] = (__bf16)(acc * z);
}

// ---------------------------------------------------------------------------
// GEMM2: out = y @ W_proj + b_proj  (f32 output)
// ---------------------------------------------------------------------------
__global__ __launch_bounds__(256) void proj_gemm(const __bf16* __restrict__ Y,
                                                 const __bf16* __restrict__ Wp,
                                                 const float* __restrict__ bias,
                                                 float* __restrict__ Out) {
    int tid  = threadIdx.x;
    int lane = tid & 31;
    int wave = tid >> 5;
    int wm   = wave & 3;
    int wn   = wave >> 2;
    int m0   = blockIdx.x * 128 + wm * 32;
    int n0   = blockIdx.y * 64  + wn * 32;
    int nt   = n0 >> 4;

    v8f acc[2][2] = {};
#pragma unroll
    for (int kt = 0; kt < 8; ++kt) {
        v16bf a0 = load_a_frag(Y, m0,      kt * 32, lane);
        v16bf a1 = load_a_frag(Y, m0 + 16, kt * 32, lane);
        v16bf b0 = load_b_frag(Wp, 8, nt,     kt, lane);
        v16bf b1 = load_b_frag(Wp, 8, nt + 1, kt, lane);
        acc[0][0] = __builtin_amdgcn_wmma_f32_16x16x32_bf16(false, a0, false, b0,
                        (short)0, acc[0][0], false, false);
        acc[0][1] = __builtin_amdgcn_wmma_f32_16x16x32_bf16(false, a0, false, b1,
                        (short)0, acc[0][1], false, false);
        acc[1][0] = __builtin_amdgcn_wmma_f32_16x16x32_bf16(false, a1, false, b0,
                        (short)0, acc[1][0], false, false);
        acc[1][1] = __builtin_amdgcn_wmma_f32_16x16x32_bf16(false, a1, false, b1,
                        (short)0, acc[1][1], false, false);
    }

#pragma unroll
    for (int i = 0; i < 2; ++i) {
#pragma unroll
        for (int j = 0; j < 2; ++j) {
            int col     = n0 + j * 16 + (lane & 15);
            int rowBase = m0 + i * 16 + ((lane >> 4) << 3);
            float bb    = bias[col];
#pragma unroll
            for (int e = 0; e < 8; ++e) {
                Out[(size_t)(rowBase + e) * C_DIM + col] = acc[i][j][e] + bb;
            }
        }
    }
}

// ---------------------------------------------------------------------------
// Launch
// ---------------------------------------------------------------------------
extern "C" void kernel_launch(void* const* d_in, const int* in_sizes, int n_in,
                              void* d_out, int out_size, void* d_ws, size_t ws_size,
                              hipStream_t stream) {
    const float* x     = (const float*)d_in[0];   // [4,20480,256]
    const float* Wqkv  = (const float*)d_in[1];   // [256,768]
    const float* Wproj = (const float*)d_in[2];   // [256,256]
    const float* bproj = (const float*)d_in[3];   // [256]

    char* ws = (char*)d_ws;
    const size_t ACT = (size_t)M_ROWS * C_DIM * 2;     // 41,943,040 bytes
    __bf16* xb      = (__bf16*)(ws);
    __bf16* q       = (__bf16*)(ws + ACT);
    __bf16* k       = (__bf16*)(ws + 2 * ACT);
    __bf16* v       = (__bf16*)(ws + 3 * ACT);
    __bf16* y       = (__bf16*)(ws + 4 * ACT);
    size_t  off     = 5 * ACT;
    __bf16* wqkv_p  = (__bf16*)(ws + off);  off += 393216;
    __bf16* wproj_p = (__bf16*)(ws + off);  off += 131072;
    float*  kv      = (float*) (ws + off);  off += 131072;
    float*  ksum    = (float*) (ws + off);  off += 4096;
    float*  kvp     = (float*) (ws + off);  off += (size_t)32 * NSEG * 1024 * 4;
    float*  ksp     = (float*) (ws + off);  off += (size_t)32 * NSEG * 32 * 4;

    const int NELEM = M_ROWS * C_DIM;                  // 20,971,520

    f32_to_bf16<<<NELEM / 4 / 256, 256, 0, stream>>>(x, xb, NELEM);
    pack_b<<<(256 * 768 + 255) / 256, 256, 0, stream>>>(Wqkv, wqkv_p, 256, 768);
    pack_b<<<(256 * 256 + 255) / 256, 256, 0, stream>>>(Wproj, wproj_p, 256, 256);
    qkv_gemm<<<dim3(M_ROWS / 128, N_QKV / 64), 256, 0, stream>>>(xb, wqkv_p, q, k, v);
    kv_partial<<<dim3(NSEG, 32), 256, 0, stream>>>(k, v, kvp, ksp);
    kv_finalize<<<32, 256, 0, stream>>>(kvp, ksp, kv, ksum);
    y_kernel<<<M_ROWS, 256, 0, stream>>>(q, kv, ksum, y);
    proj_gemm<<<dim3(M_ROWS / 128, C_DIM / 64), 256, 0, stream>>>(y, wproj_p, bproj,
                                                                  (float*)d_out);
}
